// GTLayer_87857851007456
// MI455X (gfx1250) — matrix operand order repaired and checked by reading, not compiled, not run on hardware
//
#include <hip/hip_runtime.h>
#include <hip/hip_bf16.h>

// ---------------------------------------------------------------------------
// Graph-transformer layer for MI455X (gfx1250, wave32).
// GEMMs run on V_WMMA_F32_16X16X32_BF16 (bf16 in, f32 accumulate).
// K=128 GEMMs (QKV, FFN1) are weight-stationary: 16 B-fragments pinned in
// VGPRs per wave, grid-stride over row tiles -> 0.5 loads per WMMA.
// N/K/MODE are template parameters so the store epilogue is branch-free with
// immediate-offset stores. Edge softmax/scatter uses f32 atomics; LayerNorm
// uses wave32 butterflies.
// ---------------------------------------------------------------------------

typedef __attribute__((ext_vector_type(16))) __bf16 v16bf;
typedef __attribute__((ext_vector_type(8)))  __bf16 v8bf;
typedef __attribute__((ext_vector_type(8)))  float  v8f;

#define LN_EPS 1e-5f

// ---- helpers --------------------------------------------------------------

static __device__ __forceinline__ float wave_sum32(float v) {
  #pragma unroll
  for (int o = 16; o > 0; o >>= 1) v += __shfl_xor(v, o, 32);
  return v;
}

// Order-preserving float -> uint mapping for atomicMax on signed floats.
static __device__ __forceinline__ unsigned f2ord(float f) {
  unsigned u = __float_as_uint(f);
  return (u & 0x80000000u) ? ~u : (u | 0x80000000u);
}
static __device__ __forceinline__ float ord2f(unsigned u) {
  unsigned v = (u & 0x80000000u) ? (u & 0x7fffffffu) : ~u;
  return __uint_as_float(v);
}
#define NEG_INF_ORD 0x007fffffu  // f2ord(-inf)

// ---- weight pre-pack into CDNA5 bf16 B-fragment layout --------------------
// Packed layout: idx = (((colTile*KT + kc)*32 + lane)*16 + j)
//   lane n<16 : element j = W[kc*32 + (j<8 ? j : 8+j), colTile*16 + n]
//   lane n+16 : element j = W[kc*32 + 8 + (j<8 ? j : 8+j), colTile*16 + n]
__global__ void pack_b_kernel(const float* __restrict__ W, __bf16* __restrict__ out,
                              int K, int N) {
  int idx = blockIdx.x * blockDim.x + threadIdx.x;
  if (idx >= K * N) return;
  int j = idx & 15;
  int t = idx >> 4;
  int lane = t & 31; t >>= 5;
  int KT = K >> 5;
  int kc = t % KT;
  int colTile = t / KT;
  int n  = lane & 15;
  int hf = lane >> 4;
  int kk = kc * 32 + hf * 8 + (j < 8 ? j : 8 + j);
  int nn = colTile * 16 + n;
  out[idx] = (__bf16)W[(size_t)kk * N + nn];
}

__global__ void f32_to_bf16_kernel(const float* __restrict__ in,
                                   __bf16* __restrict__ out, int n) {
  int i = blockIdx.x * blockDim.x + threadIdx.x;
  if (i < n) out[i] = (__bf16)in[i];
}

__global__ void init_attn_kernel(float* __restrict__ attn, unsigned* __restrict__ mseg,
                                 float* __restrict__ denom, int n, int nd) {
  int i = blockIdx.x * blockDim.x + threadIdx.x;
  if (i < nd) attn[i] = 0.0f;
  if (i < n) { mseg[i] = NEG_INF_ORD; denom[i] = 0.0f; }
}

// ---- fragment load helpers ------------------------------------------------

static __device__ __forceinline__ v16bf load_a_frag(const __bf16* p) {
  // 16-bit A 16x32 layout: elements 0..7 = K[hf*8 .. +7], 8..15 = K[16+hf*8 ..].
  v8bf alo = *(const v8bf*)(p);
  v8bf ahi = *(const v8bf*)(p + 16);
  v16bf a;
  #pragma unroll
  for (int i = 0; i < 8; ++i) { a[i] = alo[i]; a[8 + i] = ahi[i]; }
  return a;
}

// Branch-free tile-column store: 8 elements down one output column, with
// compile-time row stride so stores use immediate offsets.
template <int N, int MODE>
static __device__ __forceinline__ void store_ctile_col(void* Cout, int rbase, int col,
                                                       const v8f& c, float bb) {
  if (MODE == 1) {
    __bf16* p = (__bf16*)Cout + (size_t)rbase * N + col;
    #pragma unroll
    for (int i = 0; i < 8; ++i) {
      float v = c[i] + bb;
      v = v > 0.0f ? v : 0.0f;
      p[i * N] = (__bf16)v;
    }
  } else {
    float* p = (float*)Cout + (size_t)rbase * N + col;
    #pragma unroll
    for (int i = 0; i < 8; ++i) p[i * N] = c[i] + bb;
  }
}

// ---- weight-stationary WMMA GEMM for K=128 --------------------------------
// grid = (rowChunks, colGroups); each wave pins 16 B-fragments (4 kc x 4 col
// tiles = 128 VGPRs) and grid-strides over 16-row tiles doing 16 WMMAs per
// tile against only 8 A-fragment loads.
// MODE 0: store f32.  MODE 1: ReLU, store bf16.
template <int N, int MODE>
__global__ __launch_bounds__(256)
void wmma_gemm_bf16_k128_kernel(const __bf16* __restrict__ A,
                                const __bf16* __restrict__ Bp,
                                const float* __restrict__ bias,
                                void* __restrict__ Cout,
                                int nRowTiles) {
  const int K = 128, KT = 4;
  int lane = threadIdx.x & 31;
  int wave = threadIdx.x >> 5;
  int m  = lane & 15;
  int hf = lane >> 4;
  int colGroup = blockIdx.y;  // 4 column tiles per group

  // Pin the whole B panel for this column group in registers.
  v16bf b[KT * 4];
  {
    const __bf16* bbase = Bp + (size_t)lane * 16;
    #pragma unroll
    for (int j = 0; j < 4; ++j)
      #pragma unroll
      for (int kc = 0; kc < KT; ++kc)
        b[kc * 4 + j] =
            *(const v16bf*)(bbase + (((size_t)(colGroup * 4 + j) * KT + kc)) * 512);
  }

  float bb[4];
  #pragma unroll
  for (int j = 0; j < 4; ++j) bb[j] = bias[colGroup * 64 + j * 16 + m];

  int waveStride = gridDim.x * (blockDim.x >> 5);
  for (int rt = blockIdx.x * (blockDim.x >> 5) + wave; rt < nRowTiles; rt += waveStride) {
    int row0 = rt << 4;
    const __bf16* aRow = A + (size_t)(row0 + m) * K + hf * 8;

    v8f c[4];
    #pragma unroll
    for (int j = 0; j < 4; ++j)
      #pragma unroll
      for (int i = 0; i < 8; ++i) c[j][i] = 0.0f;

    #pragma unroll
    for (int kc = 0; kc < KT; ++kc) {
      v16bf a = load_a_frag(aRow + kc * 32);
      #pragma unroll
      for (int j = 0; j < 4; ++j)
        c[j] = __builtin_amdgcn_wmma_f32_16x16x32_bf16(
            false, a, false, b[kc * 4 + j], (short)0, c[j], false, false);
    }

    int rbase = row0 + hf * 8;
    #pragma unroll
    for (int j = 0; j < 4; ++j)
      store_ctile_col<N, MODE>(Cout, rbase, colGroup * 64 + j * 16 + m, c[j], bb[j]);
  }
}

// ---- generic WMMA GEMM (used for K=512 FFN2) ------------------------------
template <int K, int N, int MODE>
__global__ __launch_bounds__(256)
void wmma_gemm_bf16_kernel(const __bf16* __restrict__ A,
                           const __bf16* __restrict__ Bp,
                           const float* __restrict__ bias,
                           void* __restrict__ Cout,
                           int totalTiles) {
  const int NT = N >> 4;
  const int KT = K >> 5;
  int lane = threadIdx.x & 31;
  int wave = threadIdx.x >> 5;
  int tile = blockIdx.x * (blockDim.x >> 5) + wave;
  if (tile >= totalTiles) return;
  int rowTile = tile / NT;
  int colTile = tile - rowTile * NT;
  int row0 = rowTile << 4;

  int m  = lane & 15;
  int hf = lane >> 4;

  v8f c;
  #pragma unroll
  for (int i = 0; i < 8; ++i) c[i] = 0.0f;

  const __bf16* aRow   = A + (size_t)(row0 + m) * K + hf * 8;
  const __bf16* bpTile = Bp + ((size_t)colTile * KT) * 512 + (size_t)lane * 16;

  #pragma unroll
  for (int kc = 0; kc < KT; ++kc) {
    v16bf a = load_a_frag(aRow + kc * 32);
    v16bf b = *(const v16bf*)(bpTile + (size_t)kc * 512);
    c = __builtin_amdgcn_wmma_f32_16x16x32_bf16(
        false, a, false, b, (short)0, c, false, false);
  }

  int col = (colTile << 4) + m;
  store_ctile_col<N, MODE>(Cout, row0 + hf * 8, col, c, bias[col]);
}

// ---- edge attention -------------------------------------------------------
// One wave per edge: alpha = scale * dot(q[src], k[dst]); segment max via
// order-preserving uint atomicMax.
__global__ __launch_bounds__(256)
void edge_alpha_max_kernel(const float* __restrict__ qkv, const int* __restrict__ ei,
                           float* __restrict__ alpha, unsigned* __restrict__ mseg,
                           int nE) {
  int lane = threadIdx.x & 31;
  int wave = threadIdx.x >> 5;
  int e = blockIdx.x * (blockDim.x >> 5) + wave;
  if (e >= nE) return;
  int src = ei[e];
  int dst = ei[nE + e];
  float4 q = ((const float4*)(qkv + (size_t)src * 384))[lane];
  float4 k = ((const float4*)(qkv + (size_t)dst * 384 + 128))[lane];
  float p = q.x * k.x + q.y * k.y + q.z * k.z + q.w * k.w;
  p = wave_sum32(p);
  if (lane == 0) {
    float a = p * 0.08838834764831845f;  // 128^-0.5
    alpha[e] = a;
    atomicMax(&mseg[src], f2ord(a));
  }
}

// Thread per edge: e = exp(alpha - m[src]); denom[src] += e (in place over alpha).
__global__ void edge_expsum_kernel(float* __restrict__ alpha, const int* __restrict__ ei,
                                   const unsigned* __restrict__ mseg,
                                   float* __restrict__ denom, int nE) {
  int e = blockIdx.x * blockDim.x + threadIdx.x;
  if (e >= nE) return;
  int src = ei[e];
  float ex = __expf(alpha[e] - ord2f(mseg[src]));
  alpha[e] = ex;
  atomicAdd(&denom[src], ex);
}

// One wave per edge: attn[src] += (e/denom[src]) * v[dst]  (f32 scatter-add).
__global__ __launch_bounds__(256)
void edge_aggregate_kernel(const float* __restrict__ qkv, const int* __restrict__ ei,
                           const float* __restrict__ alpha, const float* __restrict__ denom,
                           float* __restrict__ attn, int nE) {
  int lane = threadIdx.x & 31;
  int wave = threadIdx.x >> 5;
  int e = blockIdx.x * (blockDim.x >> 5) + wave;
  if (e >= nE) return;
  int src = ei[e];
  int dst = ei[nE + e];
  float w = alpha[e] / denom[src];
  float4 v = ((const float4*)(qkv + (size_t)dst * 384 + 256))[lane];
  float* o = attn + (size_t)src * 128 + lane * 4;
  atomicAdd(o + 0, w * v.x);
  atomicAdd(o + 1, w * v.y);
  atomicAdd(o + 2, w * v.z);
  atomicAdd(o + 3, w * v.w);
}

// ---- residual + LayerNorm (wave per 128-float row) ------------------------
__global__ __launch_bounds__(256)
void residual_ln_kernel(const float* __restrict__ xa, const float* __restrict__ xb,
                        const float* __restrict__ g, const float* __restrict__ be,
                        float* __restrict__ outF, __bf16* __restrict__ outB, int n) {
  int lane = threadIdx.x & 31;
  int wave = threadIdx.x >> 5;
  int row = blockIdx.x * (blockDim.x >> 5) + wave;
  if (row >= n) return;
  float4 a = ((const float4*)(xa + (size_t)row * 128))[lane];
  float4 b = ((const float4*)(xb + (size_t)row * 128))[lane];
  float s0 = a.x + b.x, s1 = a.y + b.y, s2 = a.z + b.z, s3 = a.w + b.w;
  float mean = wave_sum32(s0 + s1 + s2 + s3) * (1.0f / 128.0f);
  float d0 = s0 - mean, d1 = s1 - mean, d2 = s2 - mean, d3 = s3 - mean;
  float var = wave_sum32(d0 * d0 + d1 * d1 + d2 * d2 + d3 * d3) * (1.0f / 128.0f);
  float rs = rsqrtf(var + LN_EPS);
  int c = lane * 4;
  float y0 = d0 * rs * g[c + 0] + be[c + 0];
  float y1 = d1 * rs * g[c + 1] + be[c + 1];
  float y2 = d2 * rs * g[c + 2] + be[c + 2];
  float y3 = d3 * rs * g[c + 3] + be[c + 3];
  float4 y = {y0, y1, y2, y3};
  ((float4*)(outF + (size_t)row * 128))[lane] = y;
  if (outB) {
    __bf16* ob = outB + (size_t)row * 128 + c;
    ob[0] = (__bf16)y0; ob[1] = (__bf16)y1; ob[2] = (__bf16)y2; ob[3] = (__bf16)y3;
  }
}

// ---------------------------------------------------------------------------

extern "C" void kernel_launch(void* const* d_in, const int* in_sizes, int n_in,
                              void* d_out, int out_size, void* d_ws, size_t ws_size,
                              hipStream_t stream) {
  (void)in_sizes; (void)n_in; (void)out_size; (void)ws_size;
  const int nN = 50000, D = 128, F = 512, nE = 800000;
  const int D3 = 3 * D;  // 384
  const int nRowTiles = nN / 16;  // 3125

  const float* x    = (const float*)d_in[0];
  const int*   ei   = (const int*)d_in[1];
  const float* Wqkv = (const float*)d_in[2];
  const float* bqkv = (const float*)d_in[3];
  const float* W1   = (const float*)d_in[4];
  const float* b1   = (const float*)d_in[5];
  const float* W2   = (const float*)d_in[6];
  const float* b2   = (const float*)d_in[7];
  const float* g1   = (const float*)d_in[8];
  const float* be1  = (const float*)d_in[9];
  const float* g2   = (const float*)d_in[10];
  const float* be2  = (const float*)d_in[11];

  char* ws = (char*)d_ws;
  size_t off = 0;
  auto alloc = [&](size_t bytes) -> void* {
    void* p = ws + off;
    off = (off + bytes + 255) & ~(size_t)255;
    return p;
  };

  __bf16*   wqkvp = (__bf16*)alloc((size_t)D * D3 * 2);
  __bf16*   w1p   = (__bf16*)alloc((size_t)D * F * 2);
  __bf16*   w2p   = (__bf16*)alloc((size_t)F * D * 2);
  __bf16*   xb    = (__bf16*)alloc((size_t)nN * D * 2);
  float*    qkv   = (float*)alloc((size_t)nN * D3 * 4);   // later reused as hb
  float*    alpha = (float*)alloc((size_t)nE * 4);
  unsigned* mseg  = (unsigned*)alloc((size_t)nN * 4);
  float*    denom = (float*)alloc((size_t)nN * 4);
  float*    attn  = (float*)alloc((size_t)nN * D * 4);    // later reused as h2
  float*    x1    = (float*)alloc((size_t)nN * D * 4);
  __bf16*   x1b   = (__bf16*)alloc((size_t)nN * D * 2);
  // Buffer reuse (lifetimes disjoint): FFN hidden overlays qkv, FFN out overlays attn.
  __bf16* hb = (__bf16*)qkv;   // nN*F bf16 = 51.2MB <= 76.8MB
  float*  h2 = attn;           // nN*D f32, same size

  // 1) pack weights to bf16 WMMA-B layout; convert x to bf16
  pack_b_kernel<<<(D * D3 + 255) / 256, 256, 0, stream>>>(Wqkv, wqkvp, D, D3);
  pack_b_kernel<<<(D * F + 255) / 256, 256, 0, stream>>>(W1, w1p, D, F);
  pack_b_kernel<<<(F * D + 255) / 256, 256, 0, stream>>>(W2, w2p, F, D);
  f32_to_bf16_kernel<<<(nN * D + 255) / 256, 256, 0, stream>>>(x, xb, nN * D);
  init_attn_kernel<<<(nN * D + 255) / 256, 256, 0, stream>>>(attn, mseg, denom, nN, nN * D);

  // 2) QKV GEMM (weight-stationary bf16 WMMA): 6 column groups of 64 cols
  {
    dim3 grid(104, D3 / 64);  // 104*8 waves grid-stride over 3125 row tiles
    wmma_gemm_bf16_k128_kernel<384, 0><<<grid, 256, 0, stream>>>(
        xb, wqkvp, bqkv, qkv, nRowTiles);
  }

  // 3) edge attention: logits + segment-max, exp + segment-sum, weighted scatter
  edge_alpha_max_kernel<<<(nE + 7) / 8, 256, 0, stream>>>(qkv, ei, alpha, mseg, nE);
  edge_expsum_kernel<<<(nE + 255) / 256, 256, 0, stream>>>(alpha, ei, mseg, denom, nE);
  edge_aggregate_kernel<<<(nE + 7) / 8, 256, 0, stream>>>(qkv, ei, alpha, denom, attn, nE);

  // 4) x1 = LN(x + attn); also emit bf16 copy for the FFN GEMM A-operand
  residual_ln_kernel<<<(nN + 7) / 8, 256, 0, stream>>>(x, attn, g1, be1, x1, x1b, nN);

  // 5) FFN: h = relu(x1 @ W1 + b1) [bf16 out, weight-stationary], then
  //    h2 = h @ W2 + b2 [f32, generic K=512 kernel]
  {
    dim3 grid1(104, F / 64);  // 8 column groups
    wmma_gemm_bf16_k128_kernel<512, 1><<<grid1, 256, 0, stream>>>(
        x1b, w1p, b1, hb, nRowTiles);
    int tiles2 = nRowTiles * (D / 16);
    wmma_gemm_bf16_kernel<512, 128, 0><<<(tiles2 + 7) / 8, 256, 0, stream>>>(
        hb, w2p, b2, h2, tiles2);
  }

  // 6) out = LN(x1 + h2)
  residual_ln_kernel<<<(nN + 7) / 8, 256, 0, stream>>>(
      x1, h2, g2, be2, (float*)d_out, (__bf16*)nullptr, nN);
}
